// SpBlock_7112465842619
// MI455X (gfx1250) — matrix operand order repaired and checked by reading, not compiled, not run on hardware
//
#include <hip/hip_runtime.h>
#include <hip/hip_bf16.h>
#include <math.h>

// ---------------------------------------------------------------------------
// ConvNeXtV2 block for MI455X (gfx1250, wave32, WMMA).
//   Kprep  : w1,w2 -> bf16 ; zero GRN sum-of-squares accumulator
//   Kbias  : b2'[c] = b2[c] + sum_f w2[c,f]*grn_b[f]   (folds GRN beta)
//   Kdwln  : 7x7 depthwise conv + channels-last LayerNorm -> bf16 A1 [M,96]
//   Kgemm1 : A1 x w1^T (WMMA bf16) + b1 + branchless GELU -> bf16 act [M,384]
//            + per-(n,f) sum-of-squares via wave-reduced f32 atomics
//            (LDS tile transpose so act stores are b128-coalesced,
//             A-fragments double-buffered across M-tiles)
//   Kgrn   : scale[n,f] = 1 + grn_g[f]*gx/(mean_f gx + eps), gx=sqrt(ss)
//   Kgemm2 : (act*scale) x w2^T (WMMA bf16) + b2' -> NCHW + residual x
//            (A-fragments double-buffered across K-chunks)
// Workspace use: ~193 MB of d_ws.
// ---------------------------------------------------------------------------

#define NIMG 64
#define CCH  96
#define FCH  384
#define HW   3136          // 56*56
#define MROW 200704        // NIMG*HW
#define WSZ  (CCH*FCH)     // 36864

typedef unsigned short u16;
typedef unsigned int   u32;

typedef __attribute__((ext_vector_type(16))) __bf16 v16bf;
typedef __attribute__((ext_vector_type(8)))  float  v8f;

struct FragB { uint4 q0, q1; };   // 32 bytes: one 16x32 bf16 WMMA operand

static __device__ __forceinline__ v16bf asfrag(FragB f) {
  return __builtin_bit_cast(v16bf, f);
}
static __device__ __forceinline__ v8f wmma_bf16(v16bf a, v16bf b, v8f c) {
  return __builtin_amdgcn_wmma_f32_16x16x32_bf16(false, a, false, b,
                                                 (short)0, c, false, false);
}
static __device__ __forceinline__ u16 f2bf(float f) {      // RNE fp32->bf16
  u32 u = __builtin_bit_cast(u32, f);
  u32 r = u + 0x7FFFu + ((u >> 16) & 1u);
  return (u16)(r >> 16);
}
// Branchless GELU: erf via Abramowitz-Stegun 7.1.26 (max err ~1.5e-7).
// v_rcp_f32 + v_exp_f32 are TRANS ops -> co-execute with the WMMA stream;
// no exec-mask divergence, no IEEE-division sequence.
static __device__ __forceinline__ float gelu_fast(float v) {
  float ax = fabsf(v) * 0.70710678118654752f;
  float t  = __builtin_amdgcn_rcpf(fmaf(0.3275911f, ax, 1.0f));
  float p  = fmaf(t, 1.061405429f, -1.453152027f);
  p = fmaf(t, p, 1.421413741f);
  p = fmaf(t, p, -0.284496736f);
  p = fmaf(t, p, 0.254829592f);
  float e  = __expf(-ax * ax);
  float er = copysignf(fmaf(-e, t * p, 1.0f), v);
  return 0.5f * v * (1.0f + er);
}
static __device__ __forceinline__ u32 scale_pk(u32 u, float s0, float s1) {
  float lo = __builtin_bit_cast(float, u << 16) * s0;
  float hi = __builtin_bit_cast(float, u & 0xFFFF0000u) * s1;
  return (u32)f2bf(lo) | ((u32)f2bf(hi) << 16);
}

// --------------------------- prep: bf16 weights + zero ss -------------------
__global__ __launch_bounds__(256) void k_prep(const float* __restrict__ w1,
                                              const float* __restrict__ w2,
                                              u16* __restrict__ w1bf,
                                              u16* __restrict__ w2bf,
                                              float* __restrict__ ss) {
  int idx = blockIdx.x * 256 + threadIdx.x;          // grid covers 98304
  if (idx < WSZ) {
    w1bf[idx] = f2bf(w1[idx]);
  } else if (idx < 2 * WSZ) {
    int i = idx - WSZ;
    w2bf[i] = f2bf(w2[i]);
  } else {
    ss[idx - 2 * WSZ] = 0.0f;                        // 64*384 entries
  }
}

// ---------------------- b2' = b2 + W2 * grn_b (fold GRN beta) ---------------
__global__ __launch_bounds__(256) void k_bias2(const float* __restrict__ w2,
                                               const float* __restrict__ b2,
                                               const float* __restrict__ grn_b,
                                               float* __restrict__ b2p) {
  int wave = threadIdx.x >> 5, lane = threadIdx.x & 31;
  int c = blockIdx.x * 8 + wave;
  if (c >= CCH) return;
  float s = 0.0f;
#pragma unroll
  for (int i = 0; i < 12; ++i) {
    int f = lane + 32 * i;
    s += w2[c * FCH + f] * grn_b[f];
  }
#pragma unroll
  for (int off = 16; off; off >>= 1) s += __shfl_xor(s, off, 32);
  if (lane == 0) b2p[c] = b2[c] + s;
}

// ------------- 7x7 depthwise conv + channels-last LayerNorm -> bf16 ---------
// One wave per pixel (bounds branches are wave-uniform -> pure SALU control);
// lane L handles channels L, L+32, L+64 so bf16 stores are coalesced 64B.
__global__ __launch_bounds__(256) void k_dwln(const float* __restrict__ x,
                                              const float* __restrict__ dw_w,
                                              const float* __restrict__ dw_b,
                                              const float* __restrict__ ln_g,
                                              const float* __restrict__ ln_b,
                                              u16* __restrict__ A1) {
  int wave = threadIdx.x >> 5, lane = threadIdx.x & 31;
  int m = blockIdx.x * 8 + wave;                 // 0..200703
  int n = m / HW, p = m - n * HW;
  int h = p / 56, w0 = p - h * 56;

  float v[3];
#pragma unroll
  for (int cc = 0; cc < 3; ++cc) {
    int c = lane + 32 * cc;
    const float* xp = x + ((long)(n * CCH + c)) * HW;
    const float* wp = dw_w + c * 49;
    float a = dw_b[c];
#pragma unroll
    for (int ky = 0; ky < 7; ++ky) {
      int yy = h + ky - 3;
      if (yy < 0 || yy >= 56) continue;
#pragma unroll
      for (int kx = 0; kx < 7; ++kx) {
        int xx = w0 + kx - 3;
        if (xx < 0 || xx >= 56) continue;
        a += xp[yy * 56 + xx] * wp[ky * 7 + kx];
      }
    }
    v[cc] = a;
  }
  float s  = v[0] + v[1] + v[2];
  float sq = v[0] * v[0] + v[1] * v[1] + v[2] * v[2];
#pragma unroll
  for (int off = 16; off; off >>= 1) {
    s  += __shfl_xor(s,  off, 32);
    sq += __shfl_xor(sq, off, 32);
  }
  float mean = s * (1.0f / 96.0f);
  float var  = sq * (1.0f / 96.0f) - mean * mean;
  float rstd = rsqrtf(var + 1e-6f);
#pragma unroll
  for (int cc = 0; cc < 3; ++cc) {
    int c = lane + 32 * cc;
    A1[m * CCH + c] = f2bf((v[cc] - mean) * rstd * ln_g[c] + ln_b[c]);
  }
}

// ---------------- GEMM1: [M,96] x [96,384] + bias + GELU + ss ---------------
// Wave = 64 rows x 64 cols (4x4 16x16 WMMA tiles, K=96 = 3 bf16 chunks).
// B fragments (12) hoisted to VGPRs and reused over 4 M-tiles; A fragments
// double-buffered so the next tile's loads drain behind the GELU epilogue.
__global__ __launch_bounds__(256) void k_gemm1(const u16* __restrict__ A1,
                                               const u16* __restrict__ Bw,
                                               const float* __restrict__ b1,
                                               float* __restrict__ ss,
                                               u16* __restrict__ act) {
  __shared__ u16 sT[8][16][64];
  int wave = threadIdx.x >> 5, lane = threadIdx.x & 31;
  int widx = blockIdx.x * 8 + wave;              // 0..18815
  int ngrp = widx % 6, mgrp = widx / 6;          // mgrp < 3136 (64 rows each)
  int hi = lane >> 4, lr = lane & 15;
  int n_img = (mgrp * 64) / HW;                  // 64 rows never cross n

  FragB bf[3][4];                                // full B panel for this wave
#pragma unroll
  for (int t = 0; t < 4; ++t) {
    int f = ngrp * 64 + t * 16 + lr;
    const u16* bp = Bw + f * CCH + hi * 8;
#pragma unroll
    for (int kc = 0; kc < 3; ++kc) {
      bf[kc][t].q0 = *(const uint4*)(bp + kc * 32);
      bf[kc][t].q1 = *(const uint4*)(bp + kc * 32 + 16);
    }
  }

  FragB af[2][3];                                // double-buffered A panel
  {
    const u16* ap = A1 + (mgrp * 64 + lr) * CCH + hi * 8;
#pragma unroll
    for (int kc = 0; kc < 3; ++kc) {
      af[0][kc].q0 = *(const uint4*)(ap + kc * 32);
      af[0][kc].q1 = *(const uint4*)(ap + kc * 32 + 16);
    }
  }

  float ssacc[4] = {0.f, 0.f, 0.f, 0.f};
#pragma unroll
  for (int mt = 0; mt < 4; ++mt) {
    int m0 = mgrp * 64 + mt * 16;
    int cb = mt & 1;
    if (mt < 3) {                                // prefetch next M-tile's A
      const u16* ap = A1 + (m0 + 16 + lr) * CCH + hi * 8;
#pragma unroll
      for (int kc = 0; kc < 3; ++kc) {
        af[cb ^ 1][kc].q0 = *(const uint4*)(ap + kc * 32);
        af[cb ^ 1][kc].q1 = *(const uint4*)(ap + kc * 32 + 16);
      }
    }
    v8f acc[4] = {};
#pragma unroll
    for (int kc = 0; kc < 3; ++kc) {
      v16bf av = asfrag(af[cb][kc]);
#pragma unroll
      for (int t = 0; t < 4; ++t)
        acc[t] = wmma_bf16(av, asfrag(bf[kc][t]), acc[t]);
    }
    // epilogue: bias + GELU, GRN ss accumulation, LDS transpose
#pragma unroll
    for (int t = 0; t < 4; ++t) {
      float bb = b1[ngrp * 64 + t * 16 + lr];
#pragma unroll
      for (int r = 0; r < 8; ++r) {
        float v = gelu_fast(acc[t][r] + bb);
        ssacc[t] += v * v;
        sT[wave][r + 8 * hi][t * 16 + lr] = f2bf(v);
      }
    }
    int row = lane >> 1;
    const u16* src = &sT[wave][row][0];
    u16* dst = act + (m0 + row) * FCH + ngrp * 64;
#pragma unroll
    for (int q = 0; q < 4; ++q) {
      int seg = (lane & 1) * 4 + q;
      *(uint4*)(dst + seg * 8) = *(const uint4*)(src + seg * 8);
    }
  }
  // wave-local reduce (lane l with l+16 share f), then one atomic per f
#pragma unroll
  for (int t = 0; t < 4; ++t) {
    float v = ssacc[t] + __shfl_xor(ssacc[t], 16, 32);
    if (hi == 0)
      atomicAdd(&ss[n_img * FCH + ngrp * 64 + t * 16 + lr], v);
  }
}

// ------------------ GRN: scale[n,f] = 1 + g[f]*gx/(mean+eps) ----------------
__global__ __launch_bounds__(384) void k_grn(const float* __restrict__ ss,
                                             const float* __restrict__ grn_g,
                                             float* __restrict__ scale) {
  __shared__ float red[384];
  int f = threadIdx.x, n = blockIdx.x;
  float gx = sqrtf(ss[n * FCH + f]);
  red[f] = gx;
  __syncthreads();
  if (f < 128) red[f] += red[f + 128] + red[f + 256];
  __syncthreads();
  for (int s2 = 64; s2 > 0; s2 >>= 1) {
    if (f < s2) red[f] += red[f + s2];
    __syncthreads();
  }
  float mean = red[0] * (1.0f / 384.0f);
  scale[n * FCH + f] = 1.0f + grn_g[f] * (gx / (mean + 1e-6f));
}

// --------- GEMM2: (act*scale) [M,384] x [384,96] + b2' -> NCHW + x ----------
// Wave = 64 rows x all 96 cols (6 tiles, K=384 = 12 bf16 chunks). GRN row
// scale applied while packing A fragments (double-buffered across K-chunks).
// Each lane's 8 accumulator entries are 8 consecutive pixels of one output
// channel -> two float4 (b128) out+residual ops per 16x16 tile.
__global__ __launch_bounds__(256) void k_gemm2(const u16* __restrict__ act,
                                               const u16* __restrict__ Bw,
                                               const float* __restrict__ b2p,
                                               const float* __restrict__ scale,
                                               const float* __restrict__ x,
                                               float* __restrict__ out) {
  int wave = threadIdx.x >> 5, lane = threadIdx.x & 31;
  int widx = blockIdx.x * 8 + wave;              // 0..3135
  int hi = lane >> 4, lr = lane & 15;
  int mbase = widx * 64;
  int n = mbase / HW, p0 = mbase - n * HW;       // 64 rows never cross n
  const float* sc = scale + n * FCH;

  for (int mt = 0; mt < 4; ++mt) {
    int m0 = mbase + mt * 16;
    const u16* ap = act + (m0 + lr) * FCH + hi * 8;
    v8f acc[6] = {};
    uint4 qa = *(const uint4*)(ap);
    uint4 qb = *(const uint4*)(ap + 16);
#pragma unroll
    for (int kc = 0; kc < 12; ++kc) {
      uint4 na, nb;
      if (kc < 11) {                             // prefetch next K-chunk
        na = *(const uint4*)(ap + (kc + 1) * 32);
        nb = *(const uint4*)(ap + (kc + 1) * 32 + 16);
      }
      int ka = kc * 32 + hi * 8;                 // f-index of qa.x low half
      qa.x = scale_pk(qa.x, sc[ka + 0], sc[ka + 1]);
      qa.y = scale_pk(qa.y, sc[ka + 2], sc[ka + 3]);
      qa.z = scale_pk(qa.z, sc[ka + 4], sc[ka + 5]);
      qa.w = scale_pk(qa.w, sc[ka + 6], sc[ka + 7]);
      qb.x = scale_pk(qb.x, sc[ka + 16], sc[ka + 17]);
      qb.y = scale_pk(qb.y, sc[ka + 18], sc[ka + 19]);
      qb.z = scale_pk(qb.z, sc[ka + 20], sc[ka + 21]);
      qb.w = scale_pk(qb.w, sc[ka + 22], sc[ka + 23]);
      FragB afr; afr.q0 = qa; afr.q1 = qb;
      v16bf av = asfrag(afr);
#pragma unroll
      for (int t = 0; t < 6; ++t) {
        const u16* bp = Bw + (t * 16 + lr) * FCH + kc * 32 + hi * 8;
        FragB bfr;
        bfr.q0 = *(const uint4*)(bp);
        bfr.q1 = *(const uint4*)(bp + 16);
        acc[t] = wmma_bf16(av, asfrag(bfr), acc[t]);
      }
      qa = na; qb = nb;
    }
    // epilogue: bias + NCHW transpose + residual, float4 stores
#pragma unroll
    for (int t = 0; t < 6; ++t) {
      int c = t * 16 + lr;
      float bb = b2p[c];
      long base = (long)(n * CCH + c) * HW + (p0 + mt * 16 + hi * 8);
      float4 x0 = *(const float4*)(x + base);
      float4 x1 = *(const float4*)(x + base + 4);
      float4 o0, o1;
      o0.x = acc[t][0] + bb + x0.x;  o0.y = acc[t][1] + bb + x0.y;
      o0.z = acc[t][2] + bb + x0.z;  o0.w = acc[t][3] + bb + x0.w;
      o1.x = acc[t][4] + bb + x1.x;  o1.y = acc[t][5] + bb + x1.y;
      o1.z = acc[t][6] + bb + x1.z;  o1.w = acc[t][7] + bb + x1.w;
      *(float4*)(out + base)     = o0;
      *(float4*)(out + base + 4) = o1;
    }
  }
}

// ---------------------------------------------------------------------------
extern "C" void kernel_launch(void* const* d_in, const int* in_sizes, int n_in,
                              void* d_out, int out_size, void* d_ws, size_t ws_size,
                              hipStream_t stream) {
  const float* x     = (const float*)d_in[0];
  const float* dw_w  = (const float*)d_in[1];
  const float* dw_b  = (const float*)d_in[2];
  const float* ln_g  = (const float*)d_in[3];
  const float* ln_b  = (const float*)d_in[4];
  const float* w1    = (const float*)d_in[5];
  const float* b1    = (const float*)d_in[6];
  const float* grn_g = (const float*)d_in[7];
  const float* grn_b = (const float*)d_in[8];
  const float* w2    = (const float*)d_in[9];
  const float* b2    = (const float*)d_in[10];
  float* out = (float*)d_out;

  // workspace carve (~193 MB)
  char* p = (char*)d_ws;
  u16*   A1    = (u16*)p;                 p += (size_t)MROW * CCH * 2;   // 38.5 MB
  u16*   actb  = (u16*)p;                 p += (size_t)MROW * FCH * 2;   // 154 MB
  u16*   w1bf  = (u16*)p;                 p += (size_t)WSZ * 2;
  u16*   w2bf  = (u16*)p;                 p += (size_t)WSZ * 2;
  float* ss    = (float*)p;               p += (size_t)NIMG * FCH * 4;
  float* scl   = (float*)p;               p += (size_t)NIMG * FCH * 4;
  float* b2p   = (float*)p;               p += 128 * 4;

  k_prep <<<384,   256, 0, stream>>>(w1, w2, w1bf, w2bf, ss);
  k_bias2<<<12,    256, 0, stream>>>(w2, b2, grn_b, b2p);
  k_dwln <<<25088, 256, 0, stream>>>(x, dw_w, dw_b, ln_g, ln_b, A1);
  k_gemm1<<<2352,  256, 0, stream>>>(A1, w1bf, b1, ss, actb);
  k_grn  <<<64,    384, 0, stream>>>(ss, grn_g, scl);
  k_gemm2<<<392,   256, 0, stream>>>(actb, w2bf, b2p, scl, x, out);
}
